// WaveletSkipConnection_48335561949949
// MI455X (gfx1250) — matrix operand order repaired and checked by reading, not compiled, not run on hardware
//
#include <hip/hip_runtime.h>

// ---------------------------------------------------------------------------
// WaveletSkipConnection on MI455X (gfx1250)
//   x: (16,256,128,128) f32, weight: (256,1024) f32
//   out[b,c,h,w] = sum_k weight[c,k] * dwt2_db2(x)[b,k,h,w],  (h,w) in 65x65
//
// K1 (DWT): block = (b, 32-ch group, h2-segment). Rolling 4-row bf16 ring in
//     LDS (symmetric ext applied at load), per-h2 band outputs staged in LDS,
//     then written as dense 512B wave-stores in WMMA-A fragment order.
//     xw (139MB bf16) is L2-resident (192MB L2); x streamed with NT loads.
// K2: weight -> bf16 (natural [c][k] == WMMA-B lane layout).
// K3 (GEMM): wave = 32M x 64N, K=1024, register double-buffered loads,
//     8 v_wmma_f32_16x16x32_bf16 per K-step, LDS-transposed NT output stores.
// ---------------------------------------------------------------------------

typedef __attribute__((ext_vector_type(16))) __bf16 v16bf;
typedef __attribute__((ext_vector_type(8)))  __bf16 v8bf;
typedef __attribute__((ext_vector_type(8)))  float  v8f;
typedef __attribute__((ext_vector_type(4)))  float  v4f;
typedef __attribute__((ext_vector_type(4)))  unsigned int v4u;

#define BATCH   16
#define CH      256
#define K4      1024
#define HO      65
#define SP      4225            // 65*65 valid m per image
#define MT2     266             // padded M tiles per image (16m each)
#define KTILES  32
#define MGROUPS 133             // MT2/2
#define NSEG    5               // h2 segments per (b,cgroup)
#define SEGROWS 13              // 65/5
#define RSTRIDE 136             // LDS row pitch (ushort), 272B = 17*16
#define SSTRIDE 40              // LDS staging kk pitch (ushort), 80B

static __device__ __forceinline__ unsigned short f2bf(float f) {
  unsigned u = __float_as_uint(f);
  return (unsigned short)((u + 0x7FFFu + ((u >> 16) & 1u)) >> 16);
}
static __device__ __forceinline__ float bf2f(unsigned short s) {
  return __uint_as_float((unsigned)s << 16);
}
// symmetric half-sample extension for n=128, pad=3 (e in [0,132])
static __device__ __forceinline__ int symidx(int e) {
  int i = e - 3;
  if (i < 0)   i = -1 - i;
  if (i > 127) i = 255 - i;
  return i;
}

// ---------------------------------------------------------------------------
// Kernel 1: DWT, 32 channels per block, coalesced fragment-order output.
// Fragment block (1KB) layout: [half(2)][lane(32)][16B]:
//   lane=(mrel,grp), half h: halfword p -> K = h*16 + grp*8 + p.
// ---------------------------------------------------------------------------
__global__ void __launch_bounds__(256)
dwt_swizzle_kernel(const float* __restrict__ x, unsigned short* __restrict__ xw)
{
  __shared__ unsigned short rows[4 * 32 * RSTRIDE];   // 34,816 B ring (by e&3)
  __shared__ unsigned short stag[4 * HO * SSTRIDE];   // 20,800 B [band][w2][kk]

  const int bid = blockIdx.x;                // 16 * 8 * NSEG
  const int b   = bid / (8 * NSEG);
  const int r0  = bid - b * (8 * NSEG);
  const int cg  = r0 / NSEG;                 // channel group (32 ch)
  const int seg = r0 - cg * NSEG;

  const int tid  = threadIdx.x;
  const int lane = tid & 31;
  const int wv   = tid >> 5;

  const int h2lo = seg * SEGROWS;
  const int h2hi = h2lo + SEGROWS;

  const float frLO[4] = { 0.48296291314469025f, 0.836516303737469f,
                          0.22414386804185735f, -0.12940952255092145f };
  const float frHI[4] = { -0.12940952255092145f, -0.22414386804185735f,
                          0.836516303737469f,   -0.48296291314469025f };

  // per-thread loader mapping: channel + 16-col chunk
  const int lch = tid >> 3;
  const int lc  = (tid & 7) * 16;
  const float* lsrc =
      x + (size_t)(b * CH + cg * 32 + lch) * (128 * 128) + lc;

  int eNext = 2 * h2lo + 1;
  for (int h2 = h2lo; h2 < h2hi; ++h2) {
    // ---- load new extended rows into ring (symmetric ext at load) ----
    while (eNext <= 2 * h2 + 4) {
      const int hr   = symidx(eNext);
      const int slot = eNext & 3;
      const float* sp = lsrc + hr * 128;
      v4f f0 = __builtin_nontemporal_load((const v4f*)(sp + 0));
      v4f f1 = __builtin_nontemporal_load((const v4f*)(sp + 4));
      v4f f2 = __builtin_nontemporal_load((const v4f*)(sp + 8));
      v4f f3 = __builtin_nontemporal_load((const v4f*)(sp + 12));
      float f[16] = { f0.x,f0.y,f0.z,f0.w, f1.x,f1.y,f1.z,f1.w,
                      f2.x,f2.y,f2.z,f2.w, f3.x,f3.y,f3.z,f3.w };
      unsigned p[8];
      #pragma unroll
      for (int i = 0; i < 8; ++i)
        p[i] = (unsigned)f2bf(f[2*i]) | ((unsigned)f2bf(f[2*i+1]) << 16);
      unsigned short* dst = rows + (size_t)(slot * 32 + lch) * RSTRIDE + lc;
      *(v4u*)(dst)     = (v4u){ p[0], p[1], p[2], p[3] };
      *(v4u*)(dst + 8) = (v4u){ p[4], p[5], p[6], p[7] };
      ++eNext;
    }
    __syncthreads();

    // ---- compute: lanes sweep w2, each wave owns kk = wv + 8q ----
    #pragma unroll
    for (int pw = 0; pw < 3; ++pw) {
      const int w2 = lane + 32 * pw;
      if (w2 < HO) {
        int cq[4];
        #pragma unroll
        for (int q = 0; q < 4; ++q) cq[q] = symidx(2 * w2 + 1 + q);
        #pragma unroll
        for (int q4 = 0; q4 < 4; ++q4) {
          const int kk = wv + 8 * q4;
          float LL = 0.f, LH = 0.f, HL = 0.f, HH = 0.f;
          #pragma unroll
          for (int r = 0; r < 4; ++r) {
            const int slot = (2 * h2 + 1 + r) & 3;
            const unsigned short* rp =
                rows + (size_t)(slot * 32 + kk) * RSTRIDE;
            float rl = 0.f, rh = 0.f;
            #pragma unroll
            for (int q = 0; q < 4; ++q) {
              const float v = bf2f(rp[cq[q]]);
              rl += frLO[q] * v;
              rh += frHI[q] * v;
            }
            LL += frLO[r] * rl;  LH += frHI[r] * rl;
            HL += frLO[r] * rh;  HH += frHI[r] * rh;
          }
          unsigned short* sb = stag + (size_t)w2 * SSTRIDE + kk;
          sb[(size_t)0 * HO * SSTRIDE] = f2bf(LL);
          sb[(size_t)1 * HO * SSTRIDE] = f2bf(LH);
          sb[(size_t)2 * HO * SSTRIDE] = f2bf(HL);
          sb[(size_t)3 * HO * SSTRIDE] = f2bf(HH);
        }
      }
    }
    __syncthreads();

    // ---- write: 5 m-tiles x 4 bands, one wave-task each, dense 512B ----
    const int mtlo = (h2 * HO) >> 4;
    for (int t = wv; t < 20; t += 8) {
      const int band = t & 3;
      const int mt   = mtlo + (t >> 2);
      const int mrel = lane & 15;
      const int grp  = lane >> 4;
      const int w2   = mt * 16 + mrel - h2 * HO;
      if (w2 >= 0 && w2 < HO) {
        const int kt = band * 8 + cg;
        char* blk = (char*)xw + ((size_t)(b * MT2 + mt) * KTILES + kt) * 1024;
        #pragma unroll
        for (int h = 0; h < 2; ++h) {
          const int kk0 = h * 16 + grp * 8;
          const v4u d = *(const v4u*)(stag
              + ((size_t)band * HO + w2) * SSTRIDE + kk0);
          *(v4u*)(blk + h * 512 + (size_t)lane * 16) = d;
        }
      }
    }
    __syncthreads();
  }
}

// ---------------------------------------------------------------------------
// Kernel 2: weight f32 -> bf16 (wb[c*1024+k])
// ---------------------------------------------------------------------------
__global__ void __launch_bounds__(256)
weight_bf16_kernel(const float* __restrict__ w, unsigned short* __restrict__ wb)
{
  const int i = blockIdx.x * 256 + threadIdx.x;
  wb[i] = f2bf(w[i]);
}

// ---------------------------------------------------------------------------
// Kernel 3: GEMM out[m,n] += A[m,k]*B[k,n];  wave = 32M x 64N, K=1024.
// ---------------------------------------------------------------------------
__global__ void __launch_bounds__(256)
wmma_gemm_kernel(const unsigned short* __restrict__ xw,
                 const unsigned short* __restrict__ wb,
                 float* __restrict__ out)
{
  __shared__ float dstage[8][4 * 16 * 20];   // 5120B per wave, 40KB total

  const int lane = threadIdx.x & 31;
  const int wv   = threadIdx.x >> 5;
  const int wid  = blockIdx.x * 8 + wv;

  const int b   = wid / (MGROUPS * 4);
  const int rem = wid - b * (MGROUPS * 4);
  const int mg  = rem >> 2;
  const int ng  = rem & 3;
  const int mt0 = mg * 2;

  const char* Ab[2];
  #pragma unroll
  for (int mi = 0; mi < 2; ++mi)
    Ab[mi] = (const char*)xw + ((size_t)(b * MT2 + mt0 + mi) * KTILES) * 1024;
  const int aLo = (lane >> 4) * 512 + (lane & 15) * 16;
  const int aHi = aLo + 256;

  const int nlo   = lane & 15;
  const int khalf = (lane >> 4) * 16;
  const char* Bb[4];
  #pragma unroll
  for (int j = 0; j < 4; ++j) {
    const int n = (ng * 4 + j) * 16 + nlo;
    Bb[j] = (const char*)wb + ((size_t)n * K4 + khalf) * 2;
  }

  v8f acc[2][4] = {};

  auto loadA = [&](const char* p) -> v16bf {
    const v8bf lo = *(const v8bf*)(p + aLo);
    const v8bf hi = *(const v8bf*)(p + aHi);
    return __builtin_shufflevector(lo, hi, 0,1,2,3,4,5,6,7,
                                           8,9,10,11,12,13,14,15);
  };

  v16bf aC[2], bC[4];
  #pragma unroll
  for (int mi = 0; mi < 2; ++mi) aC[mi] = loadA(Ab[mi]);
  #pragma unroll
  for (int j = 0; j < 4; ++j)    bC[j] = *(const v16bf*)(Bb[j]);

  #pragma unroll
  for (int kt = 0; kt < KTILES; ++kt) {
    v16bf aN[2], bN[4];
    if (kt + 1 < KTILES) {                       // prefetch next K-step
      #pragma unroll
      for (int mi = 0; mi < 2; ++mi) aN[mi] = loadA(Ab[mi] + (kt + 1) * 1024);
      #pragma unroll
      for (int j = 0; j < 4; ++j)    bN[j] = *(const v16bf*)(Bb[j] + (kt + 1) * 64);
    }
    #pragma unroll
    for (int mi = 0; mi < 2; ++mi)
      #pragma unroll
      for (int j = 0; j < 4; ++j)
        acc[mi][j] = __builtin_amdgcn_wmma_f32_16x16x32_bf16(
            false, aC[mi], false, bC[j], (short)0, acc[mi][j], false, false);
    #pragma unroll
    for (int mi = 0; mi < 2; ++mi) aC[mi] = aN[mi];
    #pragma unroll
    for (int j = 0; j < 4; ++j)    bC[j] = bN[j];
  }

  // ---- LDS-transposed, coalesced NT output stores ----
  float* ws = dstage[wv];
  const int grp = lane >> 4;
  #pragma unroll
  for (int mi = 0; mi < 2; ++mi) {
    const int mt = mt0 + mi;
    if (mt >= MT2 - 1) continue;                 // dummy padded tile
    #pragma unroll
    for (int j = 0; j < 4; ++j) {
      float* rp = ws + (j * 16 + nlo) * 20 + grp * 8;
      *(v4f*)(rp)     = __builtin_shufflevector(acc[mi][j], acc[mi][j], 0,1,2,3);
      *(v4f*)(rp + 4) = __builtin_shufflevector(acc[mi][j], acc[mi][j], 4,5,6,7);
    }
    // readback: lane -> (channel, m-half); 32B contiguous per lane
    #pragma unroll
    for (int j = 0; j < 4; ++j) {
      const int c    = ng * 64 + j * 16 + (lane >> 1);
      const int half = lane & 1;
      const float* rp = ws + (j * 16 + (lane >> 1)) * 20 + half * 8;
      const int m0 = mt * 16 + half * 8;
      float* op = out + (size_t)(b * CH + c) * SP + m0;
      if (mt < 264) {
        __builtin_nontemporal_store(*(const v4f*)(rp),     (v4f*)op);
        __builtin_nontemporal_store(*(const v4f*)(rp + 4), (v4f*)(op + 4));
      } else {
        #pragma unroll
        for (int r = 0; r < 8; ++r)
          if (m0 + r < SP) op[r] = rp[r];
      }
    }
  }
}

// ---------------------------------------------------------------------------
extern "C" void kernel_launch(void* const* d_in, const int* in_sizes, int n_in,
                              void* d_out, int out_size, void* d_ws, size_t ws_size,
                              hipStream_t stream) {
  const float* x = (const float*)d_in[0];
  const float* w = (const float*)d_in[1];
  float* out     = (float*)d_out;

  unsigned short* xw = (unsigned short*)d_ws;               // 139.5 MB
  const size_t xwBytes = (size_t)BATCH * MT2 * KTILES * 1024;
  unsigned short* wb = (unsigned short*)((char*)d_ws + xwBytes); // 512 KB

  dwt_swizzle_kernel<<<BATCH * 8 * NSEG, 256, 0, stream>>>(x, xw);
  weight_bf16_kernel<<<(CH * K4) / 256, 256, 0, stream>>>(w, wb);

  const int totalWaves = BATCH * MGROUPS * 4;               // 8512
  wmma_gemm_kernel<<<totalWaves / 8, 256, 0, stream>>>(xw, wb, out);
}
